// SSMLayer_19842748907616
// MI455X (gfx1250) — compile-verified, matrix operand order checked by reading, not run on hardware
//
#include <hip/hip_runtime.h>

// ---------------------------------------------------------------------------
// Self-similarity matrix (cosine) on MI455X / gfx1250.
//   x: [8, 2048, 512] f32  ->  ssm: [8, 2048, 2048] f32
// Pass 1: row-normalize, convert to f16, stash in d_ws (16 MB).
// Pass 2: per-batch X * X^T with V_WMMA_F32_16X16X32_F16, f32 accumulate,
//         global->LDS staging via GLOBAL_LOAD_ASYNC_TO_LDS_B128 (ASYNCcnt),
//         double-buffered LDS, no long-lived staging VGPRs (no spills).
// ---------------------------------------------------------------------------

typedef __attribute__((ext_vector_type(16))) _Float16 v16h;
typedef __attribute__((ext_vector_type(8)))  _Float16 v8h;
typedef __attribute__((ext_vector_type(4)))  _Float16 v4h;
typedef __attribute__((ext_vector_type(8)))  float    v8f;
typedef __attribute__((ext_vector_type(4)))  int      v4i;

static constexpr int NB = 8;      // batches
static constexpr int NT = 2048;   // time steps
static constexpr int ND = 512;    // feature dim

static constexpr int BM    = 128; // block tile M
static constexpr int BN    = 128; // block tile N
static constexpr int BK    = 64;  // K chunk staged through LDS
static constexpr int LDSK  = 72;  // padded LDS row stride (halves): 144B/row
                                  // row*36 dwords mod 64 tiles all banks ->
                                  // conflict-free 16-lane ds_load_b128.
static constexpr int PANEL = BM * LDSK;          // halves per panel buffer
static constexpr int NSTAGE = ND / BK;           // 8

#define AS1 __attribute__((address_space(1)))
#define AS3 __attribute__((address_space(3)))

#if defined(__has_builtin)
#if __has_builtin(__builtin_amdgcn_global_load_async_to_lds_b128)
#define HAS_ASYNC_LDS 1
#endif
#if __has_builtin(__builtin_amdgcn_s_wait_asynccnt)
#define HAS_WAIT_ASYNC 1
#endif
#endif

// 16B global -> LDS copy. Async (no VGPR transit) when the toolchain has it.
// Builtin signature (decoded from clang diagnostics):
//   void (v4i addrspace(1)* gsrc, v4i addrspace(3)* ldst, imm offset, imm cpol)
__device__ __forceinline__ void copy16_g2l(const _Float16* g, _Float16* l)
{
#if defined(HAS_ASYNC_LDS)
    __builtin_amdgcn_global_load_async_to_lds_b128(
        (AS1 v4i*)g, (AS3 v4i*)l, /*offset=*/0, /*cpol=*/0);
#else
    *(uint4*)l = *(const uint4*)g;   // fallback: load + ds_store
#endif
}

template <int N>
__device__ __forceinline__ void wait_async()
{
#if defined(HAS_ASYNC_LDS)
#if defined(HAS_WAIT_ASYNC)
    __builtin_amdgcn_s_wait_asynccnt(N);
#else
    asm volatile("s_wait_asynccnt %0" :: "i"(N) : "memory");
#endif
#endif
}

// ---------------------------------------------------------------------------
// Kernel 1: per-row L2 normalization, f32 -> f16.  One wave32 per row.
// ---------------------------------------------------------------------------
__global__ __launch_bounds__(256) void normalize_rows_f16(
    const float* __restrict__ x, _Float16* __restrict__ xn)
{
    const int lane = threadIdx.x & 31;
    const int wave = threadIdx.x >> 5;
    const int row  = blockIdx.x * 8 + wave;        // 0 .. NB*NT-1

    const float4* src = (const float4*)(x + (size_t)row * ND);
    float4 v[4];
    float ss = 0.f;
#pragma unroll
    for (int j = 0; j < 4; ++j) {                  // fully coalesced b128 loads
        v[j] = src[lane + 32 * j];
        ss += v[j].x * v[j].x + v[j].y * v[j].y + v[j].z * v[j].z + v[j].w * v[j].w;
    }
#pragma unroll
    for (int off = 16; off > 0; off >>= 1)         // wave32 reduction
        ss += __shfl_xor(ss, off, 32);

    const float scale = 1.0f / fmaxf(sqrtf(ss), 1e-12f);

    v4h* dst = (v4h*)(xn + (size_t)row * ND);
#pragma unroll
    for (int j = 0; j < 4; ++j) {
        v4h h;
        h[0] = (_Float16)(v[j].x * scale);
        h[1] = (_Float16)(v[j].y * scale);
        h[2] = (_Float16)(v[j].z * scale);
        h[3] = (_Float16)(v[j].w * scale);
        dst[lane + 32 * j] = h;                    // coalesced 8B stores
    }
}

// ---------------------------------------------------------------------------
// Fragment load from LDS implementing the CDNA5 16-bit A/B VGPR swizzle:
//   lanes 0-15 : row = lane,     comps 0-7 -> K=kb..kb+7, 8-15 -> kb+16..kb+23
//   lanes 16-31: row = lane-16,  K offsets shifted by +8
// ---------------------------------------------------------------------------
__device__ __forceinline__ v16h load_frag(const _Float16* lds, int row0,
                                          int kk, int lane)
{
    const int r  = row0 + (lane & 15);
    const int kb = kk + ((lane >> 4) << 3);
    const _Float16* p = lds + r * LDSK + kb;
    v8h lo = *(const v8h*)(p);
    v8h hi = *(const v8h*)(p + 16);
    return __builtin_shufflevector(lo, hi, 0, 1, 2, 3, 4, 5, 6, 7,
                                           8, 9, 10, 11, 12, 13, 14, 15);
}

// ---------------------------------------------------------------------------
// Kernel 2: per-batch X * X^T.
// grid = (NT/BN, NT/BM, NB), block = 256 (8 waves, 4x2 wave grid).
// Dynamic LDS: [A0 | A1 | B0 | B1], each BM*LDSK halves (72 KB total).
// ---------------------------------------------------------------------------
__global__ __launch_bounds__(256) void ssm_wmma(
    const _Float16* __restrict__ xn, float* __restrict__ out)
{
    extern __shared__ _Float16 smem[];
    _Float16* const smA = smem;                // 2 buffers
    _Float16* const smB = smem + 2 * PANEL;    // 2 buffers

    const int lane  = threadIdx.x & 31;
    const int wave  = threadIdx.x >> 5;
    const int wm    = wave & 3;                // M offset wm*32
    const int wn    = wave >> 2;               // N offset wn*64
    const int batch = blockIdx.z;
    const int m0    = blockIdx.y * BM;
    const int n0    = blockIdx.x * BN;

    const _Float16* Xb   = xn + (size_t)batch * NT * ND;
    const _Float16* Arow = Xb + (size_t)m0 * ND;
    const _Float16* Brow = Xb + (size_t)n0 * ND;

    v8f acc[2][4];
#pragma unroll
    for (int i = 0; i < 2; ++i)
#pragma unroll
        for (int j = 0; j < 4; ++j)
            acc[i][j] = (v8f){0.f, 0.f, 0.f, 0.f, 0.f, 0.f, 0.f, 0.f};

    // Per stage: each thread/lane moves 4 x 16B per panel -> exactly 8 async
    // b128 ops per wave per stage (4 for A, 4 for B); ASYNCcnt completes in
    // order, so s_wait_asynccnt<=8 with two stages in flight means the older
    // stage has fully landed in LDS.
    auto issue_stage = [&](int s, int buf) {
#pragma unroll
        for (int p = 0; p < 4; ++p) {
            const int u = threadIdx.x + p * 256;     // 128 rows x 8 chunks
            const int r = u >> 3;
            const int c = u & 7;
            const size_t goff = (size_t)r * ND + s * BK + c * 8;
            const int    loff = r * LDSK + c * 8;
            copy16_g2l(Arow + goff, smA + buf * PANEL + loff);
            copy16_g2l(Brow + goff, smB + buf * PANEL + loff);
        }
    };

    issue_stage(0, 0);
    for (int s = 0; s < NSTAGE; ++s) {
        const int cur = s & 1;
        if (s + 1 < NSTAGE) {
            issue_stage(s + 1, cur ^ 1);   // prefetch next stage (other buffer)
            wait_async<8>();               // stage s (older 8 ops) has landed
        } else {
            wait_async<0>();
        }
        __syncthreads();                   // all waves' async writes visible

        const _Float16* cA = smA + cur * PANEL;
        const _Float16* cB = smB + cur * PANEL;
#pragma unroll
        for (int kk = 0; kk < BK; kk += 32) {
            v16h afrag[2], bfrag[4];
#pragma unroll
            for (int i = 0; i < 2; ++i)
                afrag[i] = load_frag(cA, wm * 32 + i * 16, kk, lane);
#pragma unroll
            for (int j = 0; j < 4; ++j)
                bfrag[j] = load_frag(cB, wn * 64 + j * 16, kk, lane);
#pragma unroll
            for (int i = 0; i < 2; ++i)
#pragma unroll
                for (int j = 0; j < 4; ++j)
                    acc[i][j] = __builtin_amdgcn_wmma_f32_16x16x32_f16(
                        false, afrag[i], false, bfrag[j],
                        (short)0, acc[i][j], false, false);
        }
        __syncthreads();                   // readers done before s+2 overwrite
    }

    // Epilogue: C layout — VGPR r, lanes 0-15 -> (M=r, N=lane),
    //                       lanes 16-31 -> (M=8+r, N=lane-16).
    const int mlane = (lane >> 4) * 8;
    const int nlane = lane & 15;
    float* Cb = out + (size_t)batch * NT * NT;
#pragma unroll
    for (int i = 0; i < 2; ++i) {
#pragma unroll
        for (int j = 0; j < 4; ++j) {
            const int n = n0 + wn * 64 + j * 16 + nlane;
#pragma unroll
            for (int r = 0; r < 8; ++r) {
                const int m = m0 + wm * 32 + i * 16 + mlane + r;
                Cb[(size_t)m * NT + n] = acc[i][j][r];
            }
        }
    }
}

// ---------------------------------------------------------------------------
extern "C" void kernel_launch(void* const* d_in, const int* in_sizes, int n_in,
                              void* d_out, int out_size, void* d_ws, size_t ws_size,
                              hipStream_t stream)
{
    const float* x   = (const float*)d_in[0];
    float*       out = (float*)d_out;
    _Float16*    xn  = (_Float16*)d_ws;    // needs 8*2048*512*2 = 16 MB of d_ws

    // Pass 1: normalize rows -> f16 panel in workspace.
    normalize_rows_f16<<<dim3((NB * NT) / 8), dim3(256), 0, stream>>>(x, xn);

    // Pass 2: 8 batched X*X^T GEMMs on WMMA (72 KB dynamic LDS).
    dim3 grid(NT / BN, NT / BM, NB);
    const size_t lds_bytes = (size_t)4 * PANEL * sizeof(_Float16);
    ssm_wmma<<<grid, dim3(256), lds_bytes, stream>>>(xn, out);
}
